// Tactician_84112639525123
// MI455X (gfx1250) — 8-bit matrix operand order repaired and checked by reading, compiled, not run on hardware
//
#include <hip/hip_runtime.h>
#include <stdint.h>

typedef __attribute__((ext_vector_type(8))) int v8i;
typedef __attribute__((ext_vector_type(4))) int v4i;
typedef __attribute__((ext_vector_type(2))) int v2i;

#define EPSQ 1e-5f
#define BM 128
#define BN 128
#define BK 64
#define LDST 80

#if defined(__has_builtin)
#if __has_builtin(__builtin_amdgcn_global_load_async_to_lds_b128) && \
    __has_builtin(__builtin_amdgcn_s_wait_asynccnt)
#define USE_ASYNC_LDS 1
#endif
#endif
#ifndef USE_ASYNC_LDS
#define USE_ASYNC_LDS 0
#endif

#define GLOBAL_AS __attribute__((address_space(1)))
#define LDS_AS __attribute__((address_space(3)))
typedef GLOBAL_AS v4i gv4i;
typedef LDS_AS v4i lv4i;

__global__ __launch_bounds__(256) void wabssum_kernel(const float4* __restrict__ W4, long long n4,
                                                      float* __restrict__ partials) {
  __shared__ float red[256];
  long long i = (long long)blockIdx.x * 256 + threadIdx.x;
  long long stride = (long long)gridDim.x * 256;
  float s = 0.f;
  for (; i < n4; i += stride) {
    float4 v = W4[i];
    s += fabsf(v.x) + fabsf(v.y) + fabsf(v.z) + fabsf(v.w);
  }
  red[threadIdx.x] = s;
  __syncthreads();
  for (int off = 128; off > 0; off >>= 1) {
    if ((int)threadIdx.x < off) red[threadIdx.x] += red[threadIdx.x + off];
    __syncthreads();
  }
  if (threadIdx.x == 0) partials[blockIdx.x] = red[0];
}

__global__ __launch_bounds__(256) void finalsum_kernel(const float* __restrict__ p, int n,
                                                       float* __restrict__ out) {
  __shared__ float red[256];
  float s = 0.f;
  for (int i = threadIdx.x; i < n; i += 256) s += p[i];
  red[threadIdx.x] = s;
  __syncthreads();
  for (int off = 128; off > 0; off >>= 1) {
    if ((int)threadIdx.x < off) red[threadIdx.x] += red[threadIdx.x + off];
    __syncthreads();
  }
  if (threadIdx.x == 0) *out = red[0];
}

__device__ __forceinline__ int pack4_i8(float q0, float q1, float q2, float q3) {
  int b0 = (int)q0 & 255, b1 = (int)q1 & 255, b2 = (int)q2 & 255, b3 = (int)q3 & 255;
  return b0 | (b1 << 8) | (b2 << 16) | (b3 << 24);
}

__global__ __launch_bounds__(256) void wquant_kernel(const float4* __restrict__ W4, long long n4,
                                                     long long n, const float* __restrict__ sumPtr,
                                                     int* __restrict__ Q4) {
  const float alpha = (*sumPtr) / (float)n + EPSQ;
  const float inv = 1.f / alpha;
  long long i = (long long)blockIdx.x * 256 + threadIdx.x;
  long long stride = (long long)gridDim.x * 256;
  for (; i < n4; i += stride) {
    float4 w = W4[i];
    float q0 = fminf(1.f, fmaxf(-1.f, rintf(w.x * inv)));
    float q1 = fminf(1.f, fmaxf(-1.f, rintf(w.y * inv)));
    float q2 = fminf(1.f, fmaxf(-1.f, rintf(w.z * inv)));
    float q3 = fminf(1.f, fmaxf(-1.f, rintf(w.w * inv)));
    Q4[i] = pack4_i8(q0, q1, q2, q3);
  }
}

__global__ __launch_bounds__(256) void aquant_kernel(const float* __restrict__ X0,
                                                     const float* __restrict__ X1,
                                                     const float* __restrict__ X2, int Dpart,
                                                     int nparts, int* __restrict__ Q4,
                                                     float* __restrict__ scaleOut) {
  __shared__ float red[256];
  const int m = blockIdx.x;
  const int D4 = Dpart >> 2;
  const int K4 = D4 * nparts;
  float mx = 0.f;
  for (int p = 0; p < nparts; ++p) {
    const float4* src = (const float4*)(((p == 0) ? X0 : (p == 1) ? X1 : X2)) + (size_t)m * D4;
    for (int k = threadIdx.x; k < D4; k += 256) {
      float4 v = src[k];
      mx = fmaxf(mx, fmaxf(fmaxf(fabsf(v.x), fabsf(v.y)), fmaxf(fabsf(v.z), fabsf(v.w))));
    }
  }
  red[threadIdx.x] = mx;
  __syncthreads();
  for (int off = 128; off > 0; off >>= 1) {
    if ((int)threadIdx.x < off) red[threadIdx.x] = fmaxf(red[threadIdx.x], red[threadIdx.x + off]);
    __syncthreads();
  }
  const float gamma = red[0] + EPSQ;
  const float inv = 127.f / gamma;
  for (int p = 0; p < nparts; ++p) {
    const float4* src = (const float4*)(((p == 0) ? X0 : (p == 1) ? X1 : X2)) + (size_t)m * D4;
    int* dst = Q4 + (size_t)m * K4 + (size_t)p * D4;
    for (int k = threadIdx.x; k < D4; k += 256) {
      float4 v = src[k];
      float q0 = fminf(127.f, fmaxf(-128.f, rintf(v.x * inv)));
      float q1 = fminf(127.f, fmaxf(-128.f, rintf(v.y * inv)));
      float q2 = fminf(127.f, fmaxf(-128.f, rintf(v.z * inv)));
      float q3 = fminf(127.f, fmaxf(-128.f, rintf(v.w * inv)));
      dst[k] = pack4_i8(q0, q1, q2, q3);
    }
  }
  if (threadIdx.x == 0) scaleOut[m] = gamma * (1.f / 127.f);
}

template <int EPI>
__global__ __launch_bounds__(256) void gemm_iu8_kernel(
    const int8_t* __restrict__ Aq, const float* __restrict__ sA, const int8_t* __restrict__ Bq,
    const float* __restrict__ wsum, float wnumel_inv, int M, int N, int K, float* __restrict__ O,
    const float* __restrict__ R, const float* __restrict__ Y, const float* __restrict__ mask) {
  __shared__ __align__(16) int8_t sAt[2][BM * LDST];
  __shared__ __align__(16) int8_t sBt[2][BN * LDST];

  const int tid = threadIdx.x;
  const int lane = tid & 31;
  const int wave = tid >> 5;
  const int ml = lane & 15;
  const int half = lane >> 4;
  const int waveM = wave & 3;
  const int waveN = wave >> 2;

  const int m0 = blockIdx.y * BM;
  const int n0 = blockIdx.x * BN;

  const float alpha = (*wsum) * wnumel_inv + EPSQ;

  const int lrow = tid >> 2;
  const int lcol = (tid & 3) * 16;

  const int8_t* Agbase = Aq + (long long)(m0 + lrow) * K + lcol;
  const int8_t* Bgbase = Bq + (long long)(n0 + lrow) * K + lcol;
  const long long rowK = (long long)64 * K;

  v8i acc[2][4] = {};

#if USE_ASYNC_LDS
  __builtin_amdgcn_global_load_async_to_lds_b128(
      (gv4i*)(Agbase), (lv4i*)&sAt[0][lrow * LDST + lcol], 0, 0);
  __builtin_amdgcn_global_load_async_to_lds_b128(
      (gv4i*)(Agbase + rowK), (lv4i*)&sAt[0][(lrow + 64) * LDST + lcol], 0, 0);
  __builtin_amdgcn_global_load_async_to_lds_b128(
      (gv4i*)(Bgbase), (lv4i*)&sBt[0][lrow * LDST + lcol], 0, 0);
  __builtin_amdgcn_global_load_async_to_lds_b128(
      (gv4i*)(Bgbase + rowK), (lv4i*)&sBt[0][(lrow + 64) * LDST + lcol], 0, 0);
  __builtin_amdgcn_s_wait_asynccnt(0);
  __syncthreads();
#else
  v4i pa0, pa1, pb0, pb1;
  pa0 = *(const v4i*)Agbase;
  pa1 = *(const v4i*)(Agbase + rowK);
  pb0 = *(const v4i*)Bgbase;
  pb1 = *(const v4i*)(Bgbase + rowK);
  *(v4i*)&sAt[0][lrow * LDST + lcol] = pa0;
  *(v4i*)&sAt[0][(lrow + 64) * LDST + lcol] = pa1;
  *(v4i*)&sBt[0][lrow * LDST + lcol] = pb0;
  *(v4i*)&sBt[0][(lrow + 64) * LDST + lcol] = pb1;
  __syncthreads();
#endif

  int buf = 0;
  for (int k0 = 0; k0 < K; k0 += BK) {
    const int kn = k0 + BK;
    const int nb = buf ^ 1;
#if USE_ASYNC_LDS
    if (kn < K) {
      __builtin_amdgcn_global_load_async_to_lds_b128(
          (gv4i*)(Agbase + kn), (lv4i*)&sAt[nb][lrow * LDST + lcol], 0, 0);
      __builtin_amdgcn_global_load_async_to_lds_b128(
          (gv4i*)(Agbase + kn + rowK), (lv4i*)&sAt[nb][(lrow + 64) * LDST + lcol], 0, 0);
      __builtin_amdgcn_global_load_async_to_lds_b128(
          (gv4i*)(Bgbase + kn), (lv4i*)&sBt[nb][lrow * LDST + lcol], 0, 0);
      __builtin_amdgcn_global_load_async_to_lds_b128(
          (gv4i*)(Bgbase + kn + rowK), (lv4i*)&sBt[nb][(lrow + 64) * LDST + lcol], 0, 0);
    }
#else
    if (kn < K) {
      pa0 = *(const v4i*)(Agbase + kn);
      pa1 = *(const v4i*)(Agbase + kn + rowK);
      pb0 = *(const v4i*)(Bgbase + kn);
      pb1 = *(const v4i*)(Bgbase + kn + rowK);
    }
#endif

    v8i af[2];
#pragma unroll
    for (int a = 0; a < 2; ++a) {
      const int8_t* base = &sAt[buf][(waveM * 32 + a * 16 + ml) * LDST + half * 16];
#pragma unroll
      for (int p = 0; p < 4; ++p) {
        v2i d = *(const v2i*)(base + p * 8 + (p >> 1) * 16);
        af[a][2 * p] = d.x;
        af[a][2 * p + 1] = d.y;
      }
    }
    v8i bf[4];
#pragma unroll
    for (int b = 0; b < 4; ++b) {
      const int8_t* base = &sBt[buf][(waveN * 64 + b * 16 + ml) * LDST + half * 16];
      v4i q0 = *(const v4i*)(base + 0);
      v4i q1 = *(const v4i*)(base + 32);
      bf[b][0] = q0.x; bf[b][1] = q0.y; bf[b][2] = q0.z; bf[b][3] = q0.w;
      bf[b][4] = q1.x; bf[b][5] = q1.y; bf[b][6] = q1.z; bf[b][7] = q1.w;
    }

#pragma unroll
    for (int a = 0; a < 2; ++a)
#pragma unroll
      for (int b = 0; b < 4; ++b)
        acc[a][b] = __builtin_amdgcn_wmma_i32_16x16x64_iu8(true, af[a], true, bf[b], acc[a][b],
                                                           false, false);

    if (kn < K) {
#if USE_ASYNC_LDS
      __builtin_amdgcn_s_wait_asynccnt(0);
      __syncthreads();
#else
      *(v4i*)&sAt[nb][lrow * LDST + lcol] = pa0;
      *(v4i*)&sAt[nb][(lrow + 64) * LDST + lcol] = pa1;
      *(v4i*)&sBt[nb][lrow * LDST + lcol] = pb0;
      *(v4i*)&sBt[nb][(lrow + 64) * LDST + lcol] = pb1;
      __syncthreads();
#endif
      buf = nb;
    }
  }

#pragma unroll
  for (int a = 0; a < 2; ++a) {
    const int mbase = m0 + waveM * 32 + a * 16 + half * 8;
#pragma unroll
    for (int r = 0; r < 8; ++r) {
      const int m = mbase + r;
      const float rowscale = alpha * sA[m];
#pragma unroll
      for (int b = 0; b < 4; ++b) {
        const int n = n0 + waveN * 64 + b * 16 + ml;
        const long long idx = (long long)m * N + n;
        float v = (float)acc[a][b][r] * rowscale;
        if (EPI == 0) {
          O[idx] = v;
        } else if (EPI == 1) {
          O[idx] = 0.5f * v * (1.f + erff(v * 0.70710678118654752f));
        } else if (EPI == 2) {
          O[idx] = R[idx] + v;
        } else {
          O[idx] = Y[idx] + mask[m] * v;
        }
      }
    }
  }
}

extern "C" void kernel_launch(void* const* d_in, const int* in_sizes, int n_in, void* d_out,
                              int out_size, void* d_ws, size_t ws_size, hipStream_t stream) {
  (void)in_sizes; (void)n_in; (void)out_size; (void)ws_size;
  const float* x = (const float*)d_in[0];
  const float* y = (const float*)d_in[1];
  const float* z = (const float*)d_in[2];
  const float* fmask = (const float*)d_in[3];
  const float* W_in = (const float*)d_in[4];
  const float* W_fc1 = (const float*)d_in[5];
  const float* W_fc2 = (const float*)d_in[6];
  const float* W_out = (const float*)d_in[7];

  const int Mtok = 8 * 1024;
  const int D = 2048, H = 8192, L = 2;
  const int K_in = 3 * D;

  char* ws = (char*)d_ws;
  size_t cur = 0;
  auto alloc = [&](size_t bytes) -> void* {
    void* p = ws + cur;
    cur = (cur + bytes + 255) & ~(size_t)255;
    return p;
  };
  float* wsum = (float*)alloc(6 * sizeof(float));
  float* wpart = (float*)alloc(6 * 1024 * sizeof(float));
  float* sX = (float*)alloc((size_t)Mtok * 4);
  float* sH = (float*)alloc((size_t)Mtok * 4);
  float* sU = (float*)alloc((size_t)Mtok * 4);
  int8_t* Winq = (int8_t*)alloc((size_t)D * K_in);
  int8_t* W1q = (int8_t*)alloc((size_t)L * H * D);
  int8_t* W2q = (int8_t*)alloc((size_t)L * D * H);
  int8_t* Woq = (int8_t*)alloc((size_t)D * D);
  int8_t* xq = (int8_t*)alloc((size_t)Mtok * K_in);
  int8_t* hq = (int8_t*)alloc((size_t)Mtok * D);
  int8_t* uq = (int8_t*)alloc((size_t)Mtok * H);
  float* hbuf = (float*)alloc((size_t)Mtok * D * 4);
  float* ubuf = (float*)alloc((size_t)Mtok * H * 4);

  struct WT { const float* W; int8_t* Q; long long n; int slot; };
  const WT wts[6] = {
      {W_in, Winq, (long long)D * K_in, 0},
      {W_fc1, W1q, (long long)H * D, 1},
      {W_fc1 + (size_t)H * D, W1q + (size_t)H * D, (long long)H * D, 2},
      {W_fc2, W2q, (long long)D * H, 3},
      {W_fc2 + (size_t)D * H, W2q + (size_t)D * H, (long long)D * H, 4},
      {W_out, Woq, (long long)D * D, 5},
  };
  for (int i = 0; i < 6; ++i) {
    wabssum_kernel<<<1024, 256, 0, stream>>>((const float4*)wts[i].W, wts[i].n >> 2,
                                             wpart + wts[i].slot * 1024);
    finalsum_kernel<<<1, 256, 0, stream>>>(wpart + wts[i].slot * 1024, 1024, wsum + wts[i].slot);
    wquant_kernel<<<1024, 256, 0, stream>>>((const float4*)wts[i].W, wts[i].n >> 2, wts[i].n,
                                            wsum + wts[i].slot, (int*)wts[i].Q);
  }

  const dim3 blk(256);
  aquant_kernel<<<Mtok, 256, 0, stream>>>(x, y, z, D, 3, (int*)xq, sX);
  gemm_iu8_kernel<0><<<dim3(D / BN, Mtok / BM), blk, 0, stream>>>(
      xq, sX, Winq, wsum + 0, 1.f / (float)((long long)D * K_in), Mtok, D, K_in, hbuf, nullptr,
      nullptr, nullptr);
  for (int l = 0; l < L; ++l) {
    aquant_kernel<<<Mtok, 256, 0, stream>>>(hbuf, nullptr, nullptr, D, 1, (int*)hq, sH);
    gemm_iu8_kernel<1><<<dim3(H / BN, Mtok / BM), blk, 0, stream>>>(
        hq, sH, W1q + (size_t)l * H * D, wsum + 1 + l, 1.f / (float)((long long)H * D), Mtok, H, D,
        ubuf, nullptr, nullptr, nullptr);
    aquant_kernel<<<Mtok, 256, 0, stream>>>(ubuf, nullptr, nullptr, H, 1, (int*)uq, sU);
    gemm_iu8_kernel<2><<<dim3(D / BN, Mtok / BM), blk, 0, stream>>>(
        uq, sU, W2q + (size_t)l * D * H, wsum + 3 + l, 1.f / (float)((long long)D * H), Mtok, D, H,
        hbuf, hbuf, nullptr, nullptr);
  }
  aquant_kernel<<<Mtok, 256, 0, stream>>>(hbuf, nullptr, nullptr, D, 1, (int*)hq, sH);
  gemm_iu8_kernel<3><<<dim3(D / BN, Mtok / BM), blk, 0, stream>>>(
      hq, sH, Woq, wsum + 5, 1.f / (float)((long long)D * D), Mtok, D, D, (float*)d_out, nullptr, y,
      fmask);
}
